// PGNetDecoderModel_37503654429417
// MI455X (gfx1250) — compile-verified, matrix operand order checked by reading, not run on hardware
//
#include <hip/hip_runtime.h>
#include <hip/hip_bf16.h>
#include <math.h>

// ---------------------------------------------------------------------------
// PGNet decoder for MI455X (gfx1250, wave32, WMMA).
//  - Big operands (encoder_states, W_enc, W_proj, outs) pre-converted to bf16
//    once; big GEMMs are LDS-staged 128x64 macro-tile bf16 WMMA (4 WMMAs per
//    B fragment per wave, b128 LDS staging, unconditional global_prefetch).
//  - Small recurrence GEMMs (M=32): direct-load dual-A WMMA kernel.
//  - Logits written straight into d_out, softmaxed in place (saves 131 MB).
// ---------------------------------------------------------------------------

typedef __bf16 bf16_t;
typedef __attribute__((ext_vector_type(16))) __bf16 v16bf;
typedef __attribute__((ext_vector_type(8)))  __bf16 v8bf;
typedef __attribute__((ext_vector_type(8)))  float  v8f;

#define B_     32
#define T_     32
#define L_     512
#define V_     512
#define H_     512
#define E_     256
#define VOCAB_ 32000
#define OOVS_  64
#define EXT_   (VOCAB_ + OOVS_)

// ---------------- WMMA fragment helpers (ISA 7.12.2 layouts) ----------------
// slot s -> k = (s&7) | ((lane&16)>>1) | ((s&8)<<1); lane row/col = lane&15.
__device__ __forceinline__ v16bf load_frag_A(const float* A, int lda, int row0,
                                             int k0, int lane) {
    v16bf out;
    const int m   = row0 + (lane & 15);
    const int khi = (lane & 16) >> 1;
    const float* p = A + (size_t)m * lda + k0;
#pragma unroll
    for (int s = 0; s < 16; ++s) {
        const int k = (s & 7) | khi | ((s & 8) << 1);
        out[s] = (bf16_t)p[k];
    }
    return out;
}

__device__ __forceinline__ v16bf load_frag_B(const float* Bm, int ldb, int k0,
                                             int col0, int lane) {
    v16bf out;
    const int n   = col0 + (lane & 15);
    const int khi = (lane & 16) >> 1;
    const float* p = Bm + (size_t)k0 * ldb + n;
#pragma unroll
    for (int s = 0; s < 16; ++s) {
        const int k = (s & 7) | khi | ((s & 8) << 1);
        out[s] = (bf16_t)p[(size_t)k * ldb];
    }
    return out;
}

// Fragment from an LDS row of 32 bf16 k-values: slots 0..7 = k khi+0..7,
// slots 8..15 = k khi+16..23 -> two 16-byte ds_load_b128's.
__device__ __forceinline__ v16bf frag_from_lds(const bf16_t* rowp, int khi) {
    v8bf lo = *(const v8bf*)(rowp + khi);
    v8bf hi = *(const v8bf*)(rowp + khi + 16);
    v16bf o;
#pragma unroll
    for (int s = 0; s < 8; ++s) { o[s] = lo[s]; o[s + 8] = hi[s]; }
    return o;
}

__device__ __forceinline__ void pack2(bf16_t* dst, float x0, float x1) {
    union { unsigned int u; bf16_t h[2]; } pk;
    pk.h[0] = (bf16_t)x0; pk.h[1] = (bf16_t)x1;     // -> v_cvt_pk_bf16_f32
    *(unsigned int*)dst = pk.u;
}

// f32 -> bf16 bulk conversion (pairs; all sizes even)
__global__ void k_cvt_bf16(const float* __restrict__ src, bf16_t* __restrict__ dst,
                           int npairs) {
    int i = blockIdx.x * blockDim.x + threadIdx.x;
    if (i >= npairs) return;
    pack2(&dst[2 * i], src[2 * i], src[2 * i + 1]);
}

// ---------------- Big LDS-staged bf16 GEMM: C = A@B + bias ----------------
// Macro-tile 128(M) x 64(N), 256 threads = 8 waves; each wave owns a 64x16
// slab: 4 WMMAs sharing one B fragment. M mult of 128, N of 64, K of 32.
__global__ void k_gemm_big(const bf16_t* __restrict__ A, const bf16_t* __restrict__ Bm,
                           const float* __restrict__ bias, float* __restrict__ C,
                           int M, int N, int K, int lda, int ldb, int ldc) {
    __shared__ bf16_t As[128 * 32];     // As[m][k]               (8 KB)
    __shared__ bf16_t Bs[64 * 32];      // Bs[n][k] (k-transposed, 4 KB)
    const int tid  = threadIdx.x;
    const int lane = tid & 31;
    const int w    = tid >> 5;
    const int wr   = w >> 2;            // 0..1 : 64-row half
    const int wc   = w & 3;             // 0..3 : 16-col quarter
    const int nblk = N >> 6;
    const int row0 = (blockIdx.x / nblk) << 7;
    const int col0 = (blockIdx.x % nblk) << 6;

    v8f acc[4];
#pragma unroll
    for (int j = 0; j < 4; ++j) acc[j] = (v8f){0.f,0.f,0.f,0.f,0.f,0.f,0.f,0.f};

    for (int k0 = 0; k0 < K; k0 += 32) {
        __syncthreads();                                   // WAR vs prior frags
        // stage A: 128x32 bf16, 512 16-byte chunks, pure copy (coalesced)
#pragma unroll
        for (int i = 0; i < 2; ++i) {
            const int cidx = i * 256 + tid;
            const int m    = cidx >> 2;
            const int koff = (cidx & 3) << 3;
            const bf16_t* g = A + (size_t)(row0 + m) * lda + k0 + koff;
            *(v8bf*)&As[m * 32 + koff] = *(const v8bf*)g;
            __builtin_prefetch(g + 32, 0, 1);              // next K chunk
        }
        // stage B transposed: Bs[n][k] <- B[k0+k][col0+n] (coalesced along n)
#pragma unroll
        for (int i = 0; i < 4; ++i) {
            const int q  = i * 256 + tid;
            const int n  = q & 63;
            const int k2 = q >> 6;                         // 0..15
            const unsigned short* g = (const unsigned short*)Bm +
                                      (size_t)(k0 + 2 * k2) * ldb + col0 + n;
            const unsigned int pk = (unsigned int)g[0] | ((unsigned int)g[ldb] << 16);
            *(unsigned int*)&Bs[n * 32 + 2 * k2] = pk;
            __builtin_prefetch(g + (size_t)32 * ldb, 0, 1);
        }
        __syncthreads();

        const int khi   = (lane & 16) >> 1;
        const int nrel  = wc * 16 + (lane & 15);
        const int mbase = wr * 64 + (lane & 15);
        v16bf bf = frag_from_lds(&Bs[nrel * 32], khi);
#pragma unroll
        for (int j = 0; j < 4; ++j) {
            v16bf a = frag_from_lds(&As[(mbase + j * 16) * 32], khi);
            acc[j] = __builtin_amdgcn_wmma_f32_16x16x32_bf16(false, a, false, bf,
                                                             (short)0, acc[j],
                                                             false, false);
        }
    }

    const int n  = col0 + wc * 16 + (lane & 15);
    const float bv = bias ? bias[n] : 0.f;
#pragma unroll
    for (int j = 0; j < 4; ++j) {
        const int rb = row0 + wr * 64 + j * 16 + ((lane >> 4) << 3);
#pragma unroll
        for (int r = 0; r < 8; ++r)
            C[(size_t)(rb + r) * ldc + n] = acc[j][r] + bv;
    }
}

// ---------------- Small dual-A WMMA GEMM (latency path, M=32) ----------------
__global__ void k_gemm_wmma(const float* __restrict__ A1, const float* __restrict__ B1,
                            int K1, int lda1,
                            const float* __restrict__ A2, const float* __restrict__ B2,
                            int K2, int lda2,
                            const float* __restrict__ bias, float* __restrict__ C,
                            int M, int N, int ldc) {
    const int wave = threadIdx.x >> 5;
    const int lane = threadIdx.x & 31;
    const int ntile = N >> 4;
    const int tiles = (M >> 4) * ntile;
    const int tile  = blockIdx.x * (blockDim.x >> 5) + wave;
    if (tile >= tiles) return;                 // wave-uniform: EXEC all-ones below
    const int row0 = (tile / ntile) << 4;
    const int col0 = (tile % ntile) << 4;

    v8f acc = {0.f,0.f,0.f,0.f,0.f,0.f,0.f,0.f};
    for (int k0 = 0; k0 < K1; k0 += 32) {
        v16bf a = load_frag_A(A1, lda1, row0, k0, lane);
        v16bf b = load_frag_B(B1, N, k0, col0, lane);
        acc = __builtin_amdgcn_wmma_f32_16x16x32_bf16(false, a, false, b,
                                                      (short)0, acc, false, false);
    }
    if (A2) {
        for (int k0 = 0; k0 < K2; k0 += 32) {
            v16bf a = load_frag_A(A2, lda2, row0, k0, lane);
            v16bf b = load_frag_B(B2, N, k0, col0, lane);
            acc = __builtin_amdgcn_wmma_f32_16x16x32_bf16(false, a, false, b,
                                                          (short)0, acc, false, false);
        }
    }
    const int n  = col0 + (lane & 15);
    const float bv = bias ? bias[n] : 0.f;
    const int rbase = row0 + ((lane >> 4) << 3);
#pragma unroll
    for (int r = 0; r < 8; ++r)
        C[(size_t)(rbase + r) * ldc + n] = acc[r] + bv;
}

// ---------------- small kernels ----------------
__global__ void k_init(const float* __restrict__ c0, const float* __restrict__ h0,
                       float* __restrict__ c, float* __restrict__ h,
                       float* __restrict__ cov, float* __restrict__ ctx) {
    int i = blockIdx.x * blockDim.x + threadIdx.x;
    if (i < B_ * L_) cov[i] = 0.f;
    if (i < B_ * V_) ctx[i] = 0.f;
    if (i < B_ * H_) { c[i] = c0[i]; h[i] = h0[i]; }
}

__global__ void k_embed(const int* __restrict__ ids, const float* __restrict__ emb,
                        float* __restrict__ xs) {
    int i = blockIdx.x * blockDim.x + threadIdx.x;          // over T*B*E
    if (i >= T_ * B_ * E_) return;
    int e = i % E_;
    int r = i / E_;
    int b = r % B_;
    int t = r / B_;
    xs[i] = emb[(size_t)ids[b * T_ + t] * E_ + e];
}

__device__ __forceinline__ float sigf(float x) { return 1.f / (1.f + __expf(-x)); }

__global__ void k_lstm(const float* __restrict__ g, float* __restrict__ c,
                       float* __restrict__ h) {
    int i = blockIdx.x * blockDim.x + threadIdx.x;          // over B*H
    if (i >= B_ * H_) return;
    int b = i / H_, j = i % H_;
    const float* gr = g + (size_t)b * 4 * H_;
    float gi = gr[j], gf = gr[H_ + j], gg = gr[2 * H_ + j], go = gr[3 * H_ + j];
    float cn = sigf(gf) * c[i] + sigf(gi) * tanhf(gg);
    c[i] = cn;
    h[i] = sigf(go) * tanhf(cn);
}

__global__ void k_attention(const float* __restrict__ enc_f, const float* __restrict__ dec_f,
                            float* __restrict__ cov, const float* __restrict__ w_cov,
                            const float* __restrict__ v, const float* __restrict__ mask,
                            const float* __restrict__ enc_states,
                            float* __restrict__ ctx, float* __restrict__ attn_out) {
    __shared__ float sh_e[L_];
    __shared__ float red[L_];
    const int b    = blockIdx.x;
    const int tid  = threadIdx.x;
    const int wave = tid >> 5;
    const int lane = tid & 31;

    for (int l = wave; l < L_; l += 16) {
        const float cv = cov[b * L_ + l];
        const float* ef = enc_f + ((size_t)b * L_ + l) * V_;
        float s = 0.f;
        for (int a = lane; a < V_; a += 32)
            s += v[a] * tanhf(ef[a] + dec_f[b * V_ + a] + cv * w_cov[a]);
        for (int off = 16; off > 0; off >>= 1) s += __shfl_xor(s, off, 32);
        if (lane == 0) sh_e[l] = s;
    }
    __syncthreads();

    float ev = sh_e[tid];
    red[tid] = ev; __syncthreads();
    for (int s = 256; s > 0; s >>= 1) { if (tid < s) red[tid] = fmaxf(red[tid], red[tid + s]); __syncthreads(); }
    const float mx = red[0]; __syncthreads();
    float p = __expf(ev - mx) * mask[b * L_ + tid];
    red[tid] = p; __syncthreads();
    for (int s = 256; s > 0; s >>= 1) { if (tid < s) red[tid] += red[tid + s]; __syncthreads(); }
    const float attn = p / red[0];
    cov[b * L_ + tid] += attn;
    attn_out[b * L_ + tid] = attn;
    __syncthreads();
    sh_e[tid] = attn;
    __syncthreads();

    float acc = 0.f;
    const float* es = enc_states + (size_t)b * L_ * V_;
    for (int l = 0; l < L_; ++l) acc += sh_e[l] * es[(size_t)l * V_ + tid];
    ctx[b * V_ + tid] = acc;
}

__global__ void k_pgen(const float* __restrict__ ctx, const float* __restrict__ c,
                       const float* __restrict__ h, const float* __restrict__ x,
                       const float* __restrict__ Wp, const float* __restrict__ bp,
                       float* __restrict__ pg) {
    const int b = threadIdx.x >> 5;
    const int lane = threadIdx.x & 31;
    float s = 0.f;
    for (int j = lane; j < V_; j += 32) s += ctx[b * V_ + j] * Wp[j];
    for (int j = lane; j < H_; j += 32) s += c[b * H_ + j] * Wp[V_ + j];
    for (int j = lane; j < H_; j += 32) s += h[b * H_ + j] * Wp[V_ + H_ + j];
    for (int j = lane; j < E_; j += 32) s += x[b * E_ + j] * Wp[V_ + 2 * H_ + j];
    for (int off = 16; off > 0; off >>= 1) s += __shfl_xor(s, off, 32);
    if (lane == 0) pg[b] = sigf(s + bp[0]);
}

__global__ void k_softmax_scale(float* __restrict__ out, const float* __restrict__ pg) {
    __shared__ float red[256];
    const int row = blockIdx.x;
    const int tid = threadIdx.x;
    float* p = out + (size_t)row * EXT_;
    float m = -1e30f;
    for (int j = tid; j < VOCAB_; j += 256) m = fmaxf(m, p[j]);
    red[tid] = m; __syncthreads();
    for (int s = 128; s > 0; s >>= 1) { if (tid < s) red[tid] = fmaxf(red[tid], red[tid + s]); __syncthreads(); }
    m = red[0]; __syncthreads();
    float sum = 0.f;
    for (int j = tid; j < VOCAB_; j += 256) sum += __expf(p[j] - m);
    red[tid] = sum; __syncthreads();
    for (int s = 128; s > 0; s >>= 1) { if (tid < s) red[tid] += red[tid + s]; __syncthreads(); }
    const float inv = pg[row] / red[0]; __syncthreads();
    for (int j = tid; j < VOCAB_; j += 256) p[j] = __expf(p[j] - m) * inv;
    if (tid < OOVS_) p[VOCAB_ + tid] = 0.f;
}

__global__ void k_scatter(float* __restrict__ out, const float* __restrict__ attns,
                          const float* __restrict__ pg, const int* __restrict__ eev) {
    const int row = blockIdx.x;           // t*B + b
    const int b = row % B_;
    const int l = threadIdx.x;
    const float f = (1.f - pg[row]) * attns[(size_t)row * L_ + l];
    atomicAdd(&out[(size_t)row * EXT_ + eev[b * L_ + l]], f);
}

// ---------------------------------------------------------------------------
extern "C" void kernel_launch(void* const* d_in, const int* in_sizes, int n_in,
                              void* d_out, int out_size, void* d_ws, size_t ws_size,
                              hipStream_t stream) {
    const int*   dec_ids    = (const int*)d_in[0];
    const float* enc_states = (const float*)d_in[1];
    const float* mask       = (const float*)d_in[2];
    const int*   eev        = (const int*)d_in[3];
    const float* c0         = (const float*)d_in[4];
    const float* h0         = (const float*)d_in[5];
    const float* emb        = (const float*)d_in[6];
    const float* W_enc      = (const float*)d_in[7];
    const float* vvec       = (const float*)d_in[8];
    const float* w_cov      = (const float*)d_in[9];
    const float* W_dec      = (const float*)d_in[10];
    const float* b_dec      = (const float*)d_in[11];
    const float* W_x        = (const float*)d_in[12];
    const float* b_x        = (const float*)d_in[13];
    const float* Wk         = (const float*)d_in[14];
    const float* Uk         = (const float*)d_in[15];
    const float* b_lstm     = (const float*)d_in[16];
    const float* W_pgen     = (const float*)d_in[17];
    const float* b_pgen     = (const float*)d_in[18];
    const float* W_out      = (const float*)d_in[19];
    const float* b_out      = (const float*)d_in[20];
    const float* W_proj     = (const float*)d_in[21];
    const float* b_proj     = (const float*)d_in[22];
    float* dout = (float*)d_out;

    // ---- workspace layout: f32 region (~40 MB) then bf16 region (~51 MB) ----
    float* ws = (float*)d_ws;
    size_t o = 0;
    float* xs    = ws + o; o += (size_t)T_ * B_ * E_;
    float* enc_f = ws + o; o += (size_t)B_ * L_ * V_;
    float* c     = ws + o; o += (size_t)B_ * H_;
    float* h     = ws + o; o += (size_t)B_ * H_;
    float* cov   = ws + o; o += (size_t)B_ * L_;
    float* ctx   = ws + o; o += (size_t)B_ * V_;
    float* xbuf  = ws + o; o += (size_t)B_ * E_;
    float* gbuf  = ws + o; o += (size_t)B_ * 4 * H_;
    float* decf  = ws + o; o += (size_t)B_ * V_;
    float* outs  = ws + o; o += (size_t)T_ * B_ * H_;
    float* attns = ws + o; o += (size_t)T_ * B_ * L_;
    float* pgens = ws + o; o += (size_t)T_ * B_;
    bf16_t* bfr = (bf16_t*)(ws + o);
    size_t ob = 0;
    bf16_t* encbf   = bfr + ob; ob += (size_t)B_ * L_ * V_;
    bf16_t* Wencbf  = bfr + ob; ob += (size_t)V_ * V_;
    bf16_t* Wprojbf = bfr + ob; ob += (size_t)H_ * VOCAB_;
    bf16_t* outsbf  = bfr + ob; ob += (size_t)T_ * B_ * H_;

    auto cvt = [&](const float* src, bf16_t* dst, size_t n) {
        int np = (int)(n / 2);
        k_cvt_bf16<<<(np + 255) / 256, 256, 0, stream>>>(src, dst, np);
    };
    auto gemm_small = [&](const float* A1, const float* B1, int K1, int lda1,
                          const float* A2, const float* B2, int K2, int lda2,
                          const float* bias, float* C, int M, int N, int ldc) {
        int tiles  = (M >> 4) * (N >> 4);
        int blocks = (tiles + 7) / 8;
        k_gemm_wmma<<<blocks, 256, 0, stream>>>(A1, B1, K1, lda1, A2, B2, K2, lda2,
                                                bias, C, M, N, ldc);
    };
    auto gemm_big = [&](const bf16_t* A, const bf16_t* Bmat, const float* bias,
                        float* C, int M, int N, int K, int lda, int ldb, int ldc) {
        int blocks = (M >> 7) * (N >> 6);
        k_gemm_big<<<blocks, 256, 0, stream>>>(A, Bmat, bias, C, M, N, K, lda, ldb, ldc);
    };

    k_init<<<(B_ * L_ + 255) / 256, 256, 0, stream>>>(c0, h0, c, h, cov, ctx);
    k_embed<<<(T_ * B_ * E_ + 255) / 256, 256, 0, stream>>>(dec_ids, emb, xs);

    // pre-convert big GEMM operands to bf16 (once)
    cvt(enc_states, encbf, (size_t)B_ * L_ * V_);
    cvt(W_enc, Wencbf, (size_t)V_ * V_);
    cvt(W_proj, Wprojbf, (size_t)H_ * VOCAB_);

    // enc_f = encoder_states @ W_enc    [16384,512] x [512,512]
    gemm_big(encbf, Wencbf, nullptr, enc_f, B_ * L_, V_, V_, V_, V_, V_);

    for (int t = 0; t < T_; ++t) {
        const float* xt = xs + (size_t)t * B_ * E_;
        // x = [inp, ctx] @ W_x + b_x
        gemm_small(xt, W_x, E_, E_, ctx, W_x + (size_t)E_ * E_, V_, V_, b_x,
                   xbuf, B_, E_, E_);
        // g = x @ Wk + h @ Uk + b_lstm
        gemm_small(xbuf, Wk, E_, E_, h, Uk, H_, H_, b_lstm, gbuf, B_, 4 * H_, 4 * H_);
        k_lstm<<<(B_ * H_ + 255) / 256, 256, 0, stream>>>(gbuf, c, h);
        // dec_f = [c, h] @ W_dec + b_dec
        gemm_small(c, W_dec, H_, H_, h, W_dec + (size_t)H_ * V_, H_, H_, b_dec,
                   decf, B_, V_, V_);
        k_attention<<<B_, 512, 0, stream>>>(enc_f, decf, cov, w_cov, vvec, mask,
                                            enc_states, ctx,
                                            attns + (size_t)t * B_ * L_);
        k_pgen<<<1, 1024, 0, stream>>>(ctx, c, h, xbuf, W_pgen, b_pgen,
                                       pgens + (size_t)t * B_);
        // out = [h, ctx] @ W_out + b_out
        gemm_small(h, W_out, H_, H_, ctx, W_out + (size_t)H_ * H_, V_, V_, b_out,
                   outs + (size_t)t * B_ * H_, B_, H_, H_);
    }

    // projection: logits straight into d_out (ldc = EXT_), softmax in place
    cvt(outs, outsbf, (size_t)T_ * B_ * H_);
    gemm_big(outsbf, Wprojbf, b_proj, dout, T_ * B_, VOCAB_, H_, H_, VOCAB_, EXT_);
    k_softmax_scale<<<T_ * B_, 256, 0, stream>>>(dout, pgens);
    k_scatter<<<T_ * B_, 512, 0, stream>>>(dout, attns, pgens, eev);
}
